// RQVAETokenizer_50989851738236
// MI455X (gfx1250) — matrix operand order repaired and checked by reading, not compiled, not run on hardware
//
#include <hip/hip_runtime.h>
#include <hip/hip_bf16.h>

typedef __bf16 bf16_t;
typedef bf16_t v16bf __attribute__((ext_vector_type(16)));
typedef bf16_t v8bf  __attribute__((ext_vector_type(8)));
typedef float  v8f   __attribute__((ext_vector_type(8)));

union Frag16 { v16bf v; v8bf h[2]; };

#define LDSW 40  // padded LDS row stride in bf16 elements (80B, 16B-aligned)

// ---------------------------------------------------------------------------
// CDNA5 async global->LDS copy (ASYNCcnt path), 16B per lane, per-lane addrs.
// ---------------------------------------------------------------------------
__device__ __forceinline__ void async_copy_b128(unsigned lds_addr, const void* gaddr)
{
    asm volatile("global_load_async_to_lds_b128 %0, %1, off"
                 :: "v"(lds_addr), "v"(gaddr)
                 : "memory");
}
__device__ __forceinline__ void wait_asynccnt_0()
{
    asm volatile("s_wait_asynccnt 0x0" ::: "memory");
}
__device__ __forceinline__ void wait_asynccnt_6()
{
    asm volatile("s_wait_asynccnt 0x6" ::: "memory");
}

// ---------------------------------------------------------------------------
// Tiled bf16 WMMA GEMM:  out = act(alpha * (A @ Wt^T) + bias)
//   A  : [M x K] bf16 row-major (M multiple of 256)
//   Wt : [N x K] bf16 (weights pre-transposed -> contiguous B fragments)
// Block: 256 threads (8 waves), tile 256x128, BK=32, double-buffered LDS
// filled by async global->LDS copies.  Wave w -> (wm = w&3) 64-row group,
// (wn = w>>2) 64-col group => 4x4 grid of 16x16 frags, 16 WMMAs / K-step.
// NTF: non-temporal f32 stores (for write-once, never-re-read outputs).
// ---------------------------------------------------------------------------
template<bool RELU, bool NTF>
__global__ __launch_bounds__(256)
void gemm_bf16_wmma(const bf16_t* __restrict__ A, const bf16_t* __restrict__ Wt,
                    const float* __restrict__ bias, float alpha,
                    float* __restrict__ outF, bf16_t* __restrict__ outH,
                    int N, int K)
{
    __shared__ bf16_t sA[2][256 * LDSW];
    __shared__ bf16_t sB[2][128 * LDSW];

    const int tid  = threadIdx.x;
    const int lane = tid & 31;
    const int w    = tid >> 5;
    const int wm   = w & 3;    // 4 row groups x 64 rows
    const int wn   = w >> 2;   // 2 col groups x 64 cols
    const long bm  = (long)blockIdx.y * 256;
    const long bn  = (long)blockIdx.x * 128;

    const unsigned ldsA = (unsigned)(size_t)&sA[0][0];
    const unsigned ldsB = (unsigned)(size_t)&sB[0][0];

    v8f acc[4][4];
#pragma unroll
    for (int i = 0; i < 4; ++i)
#pragma unroll
        for (int j = 0; j < 4; ++j)
#pragma unroll
            for (int g = 0; g < 8; ++g) acc[i][j][g] = 0.0f;

    const int lr = tid >> 2;       // 0..63 : base row within tile
    const int lc = (tid & 3) * 8;  // 0,8,16,24 : 8-half column chunk

    // Issue one 256x32 A tile + 128x32 B tile as 6 async b128 copies / thread.
    auto issue_tile = [&](int buf, int kt) {
#pragma unroll
        for (int rr = 0; rr < 4; ++rr) {               // A: rows lr+{0,64,128,192}
            int r = lr + rr * 64;
            async_copy_b128(ldsA + (unsigned)((buf * 256 * LDSW + r * LDSW + lc) * 2),
                            &A[(bm + r) * (long)K + kt + lc]);
        }
#pragma unroll
        for (int rr = 0; rr < 2; ++rr) {               // B: rows lr+{0,64}
            int r = lr + rr * 64;
            async_copy_b128(ldsB + (unsigned)((buf * 128 * LDSW + r * LDSW + lc) * 2),
                            &Wt[(bn + r) * (long)K + kt + lc]);
        }
    };

    // 16 WMMAs on one LDS buffer.
    auto compute_tile = [&](int buf) {
        // A frags: 16x32 bf16 layout — lane<16: K 0-7/16-23, lane>=16: K 8-15/24-31
        const int klo = (lane >> 4) * 8;
        Frag16 a[4];
#pragma unroll
        for (int mf = 0; mf < 4; ++mf) {
            int r = wm * 64 + mf * 16 + (lane & 15);
            a[mf].h[0] = *(const v8bf*)&sA[buf][r * LDSW + klo];
            a[mf].h[1] = *(const v8bf*)&sA[buf][r * LDSW + klo + 16];
        }
        // B frags: 32x16 bf16 — lane<16: col=lane&15, K 0-7/16-23; lane>=16: K 8-15/24-31
        const int kb = (lane >> 4) * 8;
#pragma unroll
        for (int nf = 0; nf < 4; ++nf) {
            int n = wn * 64 + nf * 16 + (lane & 15);
            Frag16 b;
            b.h[0] = *(const v8bf*)&sB[buf][n * LDSW + kb];
            b.h[1] = *(const v8bf*)&sB[buf][n * LDSW + kb + 16];
#pragma unroll
            for (int mf = 0; mf < 4; ++mf)
                acc[mf][nf] = __builtin_amdgcn_wmma_f32_16x16x32_bf16(
                    false, a[mf].v, false, b.v, (short)0, acc[mf][nf],
                    false, false);
        }
    };

    const int nsteps = K >> 5;

    // Prologue: fill buffer 0.
    issue_tile(0, 0);

    // Steady state: issue next while computing current (no branches inside).
    for (int s = 0; s < nsteps - 1; ++s) {
        const int buf = s & 1;
        issue_tile(buf ^ 1, (s + 1) * 32);
        wait_asynccnt_6();       // 6 newest (next tile) may remain in flight
        __syncthreads();         // all waves' current-buffer writes visible
        compute_tile(buf);
        __syncthreads();         // all done reading buf before it is refilled
    }

    // Epilogue: last tile.
    wait_asynccnt_0();
    __syncthreads();
    compute_tile((nsteps - 1) & 1);

    // Store: 16x16 f32 C layout — VGPR g: lanes 0-15 -> M=g, lanes 16-31 -> M=8+g
    float bv[4];
#pragma unroll
    for (int nf = 0; nf < 4; ++nf)
        bv[nf] = bias[bn + wn * 64 + nf * 16 + (lane & 15)];

#pragma unroll
    for (int mf = 0; mf < 4; ++mf) {
        long row0 = bm + wm * 64 + mf * 16 + (lane >> 4) * 8;
#pragma unroll
        for (int nf = 0; nf < 4; ++nf) {
            long col = bn + wn * 64 + nf * 16 + (lane & 15);
#pragma unroll
            for (int g = 0; g < 8; ++g) {
                float v = alpha * acc[mf][nf][g] + bv[nf];
                if (RELU) v = v > 0.0f ? v : 0.0f;
                long idx = (row0 + g) * (long)N + col;
                if (outF) {
                    if (NTF) __builtin_nontemporal_store(v, &outF[idx]);
                    else     outF[idx] = v;
                }
                if (outH) outH[idx] = (bf16_t)v;
            }
        }
    }
}

// ---------------------------------------------------------------------------
// Elementwise f32 -> bf16 convert (NT load: source is read exactly once)
// ---------------------------------------------------------------------------
__global__ __launch_bounds__(256)
void k_f32_to_bf16(const float* __restrict__ s, bf16_t* __restrict__ d, long n)
{
    long i = (long)blockIdx.x * 256 + threadIdx.x;
    if (i < n) d[i] = (bf16_t)__builtin_nontemporal_load(&s[i]);
}

// Transpose-convert: s[K][N] f32 -> d[N][K] bf16
__global__ __launch_bounds__(256)
void k_transpose_bf16(const float* __restrict__ s, bf16_t* __restrict__ d,
                      int K, int N)
{
    long i = (long)blockIdx.x * 256 + threadIdx.x;
    if (i < (long)K * N) {
        int k = (int)(i / N);
        int n = (int)(i % N);
        d[(long)n * K + k] = (bf16_t)s[i];
    }
}

// Per-code squared norms: cb [3*256][128] f32 -> cn [3*256] f32
__global__ __launch_bounds__(256)
void k_cnorm(const float* __restrict__ cb, float* __restrict__ cn)
{
    int i = blockIdx.x * 256 + threadIdx.x;
    if (i < 3 * 256) {
        const float* c = cb + (long)i * 128;
        float s = 0.f;
#pragma unroll 4
        for (int d = 0; d < 128; ++d) { float v = c[d]; s += v * v; }
        cn[i] = s;
    }
}

__global__ void k_zero1(float* p)
{
    if (threadIdx.x == 0 && blockIdx.x == 0) *p = 0.f;
}

// ---------------------------------------------------------------------------
// Per-row argmin over 256 scores + residual/quantized update (one wave / row)
// scores[b][c] = ||c||^2 - 2 r.c  (||r||^2 constant per row -> same argmin)
// ---------------------------------------------------------------------------
__global__ __launch_bounds__(256)
void rq_argmin_update(const float* __restrict__ scores,
                      const float* __restrict__ cbF,      // level's f32 codebook [256][128]
                      const float* __restrict__ residIn,  // f32 [B][128]
                      float* __restrict__ residOut,       // f32 [B][128]
                      bf16_t* __restrict__ residH,        // bf16 [B][128]
                      float* __restrict__ qacc,           // f32 [B][128]
                      bf16_t* __restrict__ qH,            // bf16 [B][128] or null
                      int* __restrict__ codes,            // [B][3]
                      float* __restrict__ loss,
                      int lvl, long B)
{
    const int lane = threadIdx.x & 31;
    const int wid  = threadIdx.x >> 5;
    const long b   = (long)blockIdx.x * 8 + wid;
    if (b >= B) return;

    const float* s = scores + b * 256;
    float best = 3.402823466e38f;
    int   bi   = 0;
#pragma unroll
    for (int j = 0; j < 8; ++j) {
        int   idx = lane + 32 * j;
        float v   = s[idx];
        if (v < best) { best = v; bi = idx; }
    }
    // wave32 reduction, lowest index wins ties (argmin semantics)
    for (int off = 16; off > 0; off >>= 1) {
        float ov = __shfl_xor(best, off, 32);
        int   oi = __shfl_xor(bi,   off, 32);
        if (ov < best || (ov == best && oi < bi)) { best = ov; bi = oi; }
    }

    const float* e = cbF + (long)bi * 128;
    float lsum = 0.f;
#pragma unroll
    for (int j = 0; j < 4; ++j) {
        int   d    = lane * 4 + j;
        float ev   = e[d];
        float rv   = residIn[b * 128 + d];
        float diff = rv - ev;               // new residual; also commit diff
        lsum += diff * diff;
        residOut[b * 128 + d] = diff;
        residH[b * 128 + d]   = (bf16_t)diff;
        float q = (lvl == 0) ? ev : (qacc[b * 128 + d] + ev);
        qacc[b * 128 + d] = q;
        if (qH) qH[b * 128 + d] = (bf16_t)q;  // straight-through value: sum of e_k
    }
    for (int off = 16; off > 0; off >>= 1) lsum += __shfl_xor(lsum, off, 32);
    if (lane == 0) {
        codes[b * 3 + lvl] = bi;
        atomicAdd(loss, lsum * (0.25f / ((float)B * 128.0f)));
    }
}

// ---------------------------------------------------------------------------
extern "C" void kernel_launch(void* const* d_in, const int* in_sizes, int n_in,
                              void* d_out, int out_size, void* d_ws, size_t ws_size,
                              hipStream_t stream)
{
    (void)in_sizes; (void)n_in; (void)out_size; (void)ws_size;
    const float* x   = (const float*)d_in[0];
    const float* ew1 = (const float*)d_in[1];
    const float* eb1 = (const float*)d_in[2];
    const float* ew2 = (const float*)d_in[3];
    const float* eb2 = (const float*)d_in[4];
    const float* ew3 = (const float*)d_in[5];
    const float* eb3 = (const float*)d_in[6];
    const float* dw1 = (const float*)d_in[7];
    const float* db1 = (const float*)d_in[8];
    const float* dw2 = (const float*)d_in[9];
    const float* db2 = (const float*)d_in[10];
    const float* dw3 = (const float*)d_in[11];
    const float* db3 = (const float*)d_in[12];
    const float* cb  = (const float*)d_in[13];

    const long B = 65536, DIN = 768, H = 512, D = 128, K = 256, L = 3;

    char* ws = (char*)d_ws;
    size_t off = 0;
    auto alloc = [&](size_t bytes) {
        size_t r = off; off += (bytes + 255) & ~(size_t)255; return r;
    };
    // Buffers (lifetime-based aliasing to cap footprint ~370 MB):
    size_t o_xh  = alloc(B * DIN * 2);  // x bf16; dead after enc stage1 -> reused for scores
    size_t o_h1  = alloc(B * H * 2);    // enc h1; reused as dec d1
    size_t o_h2  = alloc(B * H * 2);    // enc h2; reused as dec d2
    size_t o_zf  = alloc(B * D * 4);    // z f32 (residual level 0)
    size_t o_zh  = alloc(B * D * 2);    // residual bf16 (score-GEMM A input)
    size_t o_rf  = alloc(B * D * 4);    // residual f32 (levels >= 1)
    size_t o_qa  = alloc(B * D * 4);    // quantized accumulator f32
    size_t o_qh  = alloc(B * D * 2);    // quantized bf16 (decoder A input)
    size_t o_w1t = alloc(DIN * H * 2);
    size_t o_w2t = alloc(H * H * 2);
    size_t o_w3t = alloc(H * D * 2);
    size_t o_v1t = alloc(D * H * 2);
    size_t o_v2t = alloc(H * H * 2);
    size_t o_v3t = alloc(H * DIN * 2);
    size_t o_cbh = alloc(L * K * D * 2);
    size_t o_cn  = alloc(L * K * 4);

    bf16_t* xh  = (bf16_t*)(ws + o_xh);
    float*  sc  = (float*)(ws + o_xh);   // scores alias (B*256*4 <= B*768*2)
    bf16_t* h1  = (bf16_t*)(ws + o_h1);
    bf16_t* h2  = (bf16_t*)(ws + o_h2);
    float*  zf  = (float*)(ws + o_zf);
    bf16_t* zh  = (bf16_t*)(ws + o_zh);
    float*  rf  = (float*)(ws + o_rf);
    float*  qa  = (float*)(ws + o_qa);
    bf16_t* qh  = (bf16_t*)(ws + o_qh);
    bf16_t* w1t = (bf16_t*)(ws + o_w1t);
    bf16_t* w2t = (bf16_t*)(ws + o_w2t);
    bf16_t* w3t = (bf16_t*)(ws + o_w3t);
    bf16_t* v1t = (bf16_t*)(ws + o_v1t);
    bf16_t* v2t = (bf16_t*)(ws + o_v2t);
    bf16_t* v3t = (bf16_t*)(ws + o_v3t);
    bf16_t* cbh = (bf16_t*)(ws + o_cbh);
    float*  cn  = (float*)(ws + o_cn);

    float* recon = (float*)d_out;                       // [B][768] f32
    int*   codes = (int*)((float*)d_out + B * DIN);     // [B][3] i32
    float* loss  = (float*)d_out + B * DIN + B * 3;     // scalar f32

    auto tr = [&](const float* s, bf16_t* d, int Kd, int Nd) {
        long n = (long)Kd * Nd;
        k_transpose_bf16<<<dim3((unsigned)((n + 255) / 256)), 256, 0, stream>>>(s, d, Kd, Nd);
    };

    // --- Prep: converts, transposes, code norms, loss init -----------------
    k_zero1<<<1, 64, 0, stream>>>(loss);
    {
        long n = B * DIN;
        k_f32_to_bf16<<<dim3((unsigned)((n + 255) / 256)), 256, 0, stream>>>(x, xh, n);
    }
    tr(ew1, w1t, (int)DIN, (int)H);
    tr(ew2, w2t, (int)H,   (int)H);
    tr(ew3, w3t, (int)H,   (int)D);
    tr(dw1, v1t, (int)D,   (int)H);
    tr(dw2, v2t, (int)H,   (int)H);
    tr(dw3, v3t, (int)H,   (int)DIN);
    {
        long n = L * K * D;  // codebooks already [code][dim] == Wt layout
        k_f32_to_bf16<<<dim3((unsigned)((n + 255) / 256)), 256, 0, stream>>>(cb, cbh, n);
    }
    k_cnorm<<<3, 256, 0, stream>>>(cb, cn);

    const unsigned GM = (unsigned)(B / 256);

    // --- Encoder -----------------------------------------------------------
    gemm_bf16_wmma<true, false><<<dim3((unsigned)(H / 128), GM), 256, 0, stream>>>(
        xh, w1t, eb1, 1.0f, nullptr, h1, (int)H, (int)DIN);
    gemm_bf16_wmma<true, false><<<dim3((unsigned)(H / 128), GM), 256, 0, stream>>>(
        h1, w2t, eb2, 1.0f, nullptr, h2, (int)H, (int)H);
    gemm_bf16_wmma<false, false><<<dim3((unsigned)(D / 128), GM), 256, 0, stream>>>(
        h2, w3t, eb3, 1.0f, zf, zh, (int)D, (int)H);

    // --- Residual quantization: 3 levels -----------------------------------
    for (int lvl = 0; lvl < 3; ++lvl) {
        // scores = ||c||^2 - 2 * r @ c^T  via WMMA (alpha=-2, bias=cnorm)
        gemm_bf16_wmma<false, false><<<dim3((unsigned)(K / 128), GM), 256, 0, stream>>>(
            zh, cbh + (long)lvl * K * D, cn + (long)lvl * K, -2.0f,
            sc, nullptr, (int)K, (int)D);
        rq_argmin_update<<<dim3((unsigned)(B / 8)), 256, 0, stream>>>(
            sc, cb + (long)lvl * K * D,
            (lvl == 0) ? zf : rf, rf, zh, qa,
            (lvl == 2) ? qh : nullptr, codes, loss, lvl, B);
    }

    // --- Decoder (d1 aliases h1, d2 aliases h2) -----------------------------
    gemm_bf16_wmma<true, false><<<dim3((unsigned)(H / 128), GM), 256, 0, stream>>>(
        qh, v1t, db1, 1.0f, nullptr, h1, (int)H, (int)D);
    gemm_bf16_wmma<true, false><<<dim3((unsigned)(H / 128), GM), 256, 0, stream>>>(
        h1, v2t, db2, 1.0f, nullptr, h2, (int)H, (int)H);
    // recon: written once, never re-read -> non-temporal stores
    gemm_bf16_wmma<false, true><<<dim3((unsigned)(DIN / 128), GM), 256, 0, stream>>>(
        h2, v3t, db3, 1.0f, recon, nullptr, (int)DIN, (int)H);
}
